// LAHNLoss_69861938037087
// MI455X (gfx1250) — matrix operand order repaired and checked by reading, not compiled, not run on hardware
//
#include <hip/hip_runtime.h>
#include <math.h>

// ---------------- problem constants (match reference) ----------------
#define NB 2048      // BATCH
#define NE 768       // EMB_D
#define NP 128       // PROJ_D
#define NQ 32768     // QUEUE
#define NK 512       // HARD_K
#define INV_TEMP (1.0f / 0.07f)

// ---------------- WMMA / TDM types ----------------
typedef __attribute__((ext_vector_type(16))) __bf16 v16bf;
typedef __attribute__((ext_vector_type(8)))  float  v8f;
typedef unsigned int v4u  __attribute__((ext_vector_type(4)));
typedef int          v8i_ __attribute__((ext_vector_type(8)));
typedef int          v4i_ __attribute__((ext_vector_type(4)));

union FragAB { v16bf v; unsigned int u[8]; };

// ---------------- helpers ----------------
static __device__ __forceinline__ unsigned short f32_to_bf16(float f) {
  unsigned int u = __float_as_uint(f);
  unsigned int r = (u + 0x7FFFu + ((u >> 16) & 1u)) >> 16;   // round-to-nearest-even
  return (unsigned short)r;
}

static __device__ __forceinline__ unsigned int fkey(float f) {
  unsigned int u = __float_as_uint(f);
  return (u & 0x80000000u) ? ~u : (u | 0x80000000u);         // order-preserving map
}
static __device__ __forceinline__ float funkey(unsigned int k) {
  unsigned int u = (k & 0x80000000u) ? (k & 0x7FFFFFFFu) : ~k;
  return __uint_as_float(u);
}

static __device__ __forceinline__ float logaddexpf_(float a, float b) {
  float m = fmaxf(a, b), n = fminf(a, b);
  if (m == -INFINITY) return -INFINITY;
  return m + log1pf(expf(n - m));
}

// LDS byte offset of a __shared__ object (AS(3) pointer value == LDS offset)
typedef __attribute__((address_space(3))) unsigned char lds_byte;
static __device__ __forceinline__ unsigned lds_off_of(const void* p) {
  return (unsigned)(unsigned long long)(lds_byte*)p;
}

// ---------------- Tensor Data Mover: 2-D tile load global -> LDS ----------------
// D# per CDNA5 ISA: group0 {count=1, lds_addr, global_addr, type=2},
// group1 {data_size=4B, tensor_dim0/1, tile_dim0/1, dim0_stride}. Groups 2/3 unused (2-D).
// clang-23 toolchain: 6-arg builtin (g0 v4u, g1 v8i, g2 v4i, g3 v4i, pad v8i, cpol i32).
static __device__ __forceinline__ void tdm_load_2d(unsigned lds_off, const void* gaddr,
                                                   unsigned tile_w_u32, unsigned tile_h,
                                                   unsigned tensor_w_u32, unsigned tensor_h,
                                                   unsigned stride_u32) {
  unsigned long long ga = (unsigned long long)gaddr;
  v4u g0;
  g0[0] = 1u;                                                   // count=1 (valid user D#)
  g0[1] = lds_off;                                              // lds_addr (bytes)
  g0[2] = (unsigned)(ga & 0xFFFFFFFFu);                         // global_addr[31:0]
  g0[3] = (unsigned)((ga >> 32) & 0x01FFFFFFu) | (2u << 30);    // addr[56:32] | type=2
  v8i_ g1;
  g1[0] = (int)(2u << 16);                                      // data_size code 2 = 4 bytes
  g1[1] = (int)((tensor_w_u32 & 0xFFFFu) << 16);                // tensor_dim0[15:0]
  g1[2] = (int)((tensor_w_u32 >> 16) | ((tensor_h & 0xFFFFu) << 16)); // dim0[31:16]|dim1[15:0]
  g1[3] = (int)((tensor_h >> 16) | (tile_w_u32 << 16));         // dim1[31:16] | tile_dim0
  g1[4] = (int)tile_h;                                          // tile_dim1 (tile_dim2=0)
  g1[5] = (int)stride_u32;                                      // tensor_dim0_stride[31:0]
  g1[6] = 0;                                                    // stride0 hi | stride1 lo
  g1[7] = 0;
  v4i_ z4 = {0, 0, 0, 0};
  v8i_ z8 = {0, 0, 0, 0, 0, 0, 0, 0};
  __builtin_amdgcn_tensor_load_to_lds(g0, g1, z4, z4, z8, 0);
}

// ---------------- conversion / packing kernels ----------------
__global__ void __launch_bounds__(256)
cvt_f32_bf16(const float* __restrict__ in, unsigned short* __restrict__ out, int n) {
  int i = blockIdx.x * 256 + threadIdx.x;
  if (i < n) out[i] = f32_to_bf16(in[i]);
}

// B (KxN f32 row-major) -> panel: out[n*(K/2)+kp] = pack(bf16(B[2kp][n]), bf16(B[2kp+1][n]))
__global__ void __launch_bounds__(256)
pack_B_swz(const float* __restrict__ in, unsigned int* __restrict__ out, int K, int N) {
  int i = blockIdx.x * 256 + threadIdx.x;
  int Kp = K >> 1;
  if (i < N * Kp) {
    int n = i / Kp, kp = i - n * Kp;
    unsigned lo = f32_to_bf16(in[(size_t)(2 * kp) * N + n]);
    unsigned hi = f32_to_bf16(in[(size_t)(2 * kp + 1) * N + n]);
    out[i] = lo | (hi << 16);
  }
}

// src (N x K f32 row-major), B = src^T -> panel: out[n*(K/2)+kp] = pack(src[n][2kp], src[n][2kp+1])
__global__ void __launch_bounds__(256)
pack_Bt_swz(const float* __restrict__ in, unsigned int* __restrict__ out, int N, int K) {
  int i = blockIdx.x * 256 + threadIdx.x;
  int Kp = K >> 1;
  if (i < N * Kp) {
    int n = i / Kp, kp = i - n * Kp;
    const float2 v = *(const float2*)(in + (size_t)n * K + 2 * kp);
    out[i] = (unsigned)f32_to_bf16(v.x) | ((unsigned)f32_to_bf16(v.y) << 16);
  }
}

// ---------------- bf16 WMMA GEMM with TDM double-buffered staging ----------------
// A: MxK bf16 row-major. Bs: N x (K/2) u32 panel (WMMA-B fragment order).
// Block = 256 threads = 8 waves; block tile 128(M) x 64(N); wave tile 16 x 64 (4 WMMAs).
// M%128==0, N%64==0, K%32==0.
__global__ void __launch_bounds__(256)
gemm_bf16_wmma(const unsigned short* __restrict__ Ap,
               const unsigned int*   __restrict__ Bs,
               const float* __restrict__ bias,
               float* __restrict__ Cf,            // used if out_bf16 == 0
               unsigned short* __restrict__ Cb,   // used if out_bf16 == 1
               int M, int N, int K,
               float scale, int relu, int out_bf16) {
  __shared__ unsigned int lds_a[2][128 * 16];  // 128 rows x 32 bf16 (16 u32 pairs)
  __shared__ unsigned int lds_b[2][64 * 16];   // 64 n-rows x 16 kp u32

  const int tid  = threadIdx.x;
  const int wave = tid >> 5;
  const int lane = tid & 31;
  const int half = lane >> 4;
  const int r    = lane & 15;
  const int bm   = blockIdx.y * 128;
  const int bn   = blockIdx.x * 64;
  const int Kp   = K >> 1;

  v8f acc[4];
#pragma unroll
  for (int nt = 0; nt < 4; ++nt) acc[nt] = (v8f){0.f, 0.f, 0.f, 0.f, 0.f, 0.f, 0.f, 0.f};

  const int nsteps = K >> 5;   // K-steps of 32

  // prologue: DMA buffer 0 (A tile: 128 x 16 u32; B tile: 64 x 16 u32)
  if (wave == 0) {
    tdm_load_2d(lds_off_of(&lds_a[0][0]), Ap + (size_t)bm * K,
                16u, 128u, (unsigned)Kp, (unsigned)M, (unsigned)Kp);
    tdm_load_2d(lds_off_of(&lds_b[0][0]), Bs + (size_t)bn * Kp,
                16u, 64u, (unsigned)Kp, (unsigned)N, (unsigned)Kp);
  }

  for (int s = 0; s < nsteps; ++s) {
    const int cur = s & 1;
    const int nxt = cur ^ 1;
    if (wave == 0) {
      if (s + 1 < nsteps) {
        const int kp0 = (s + 1) << 4;    // u32-pair offset of next K-step
        tdm_load_2d(lds_off_of(&lds_a[nxt][0]), Ap + (size_t)bm * K + (size_t)(kp0 << 1),
                    16u, 128u, (unsigned)Kp, (unsigned)M, (unsigned)Kp);
        tdm_load_2d(lds_off_of(&lds_b[nxt][0]), Bs + (size_t)bn * Kp + (size_t)kp0,
                    16u, 64u, (unsigned)Kp, (unsigned)N, (unsigned)Kp);
        __builtin_amdgcn_s_wait_tensorcnt(2);   // current buffer's 2 DMAs complete
      } else {
        __builtin_amdgcn_s_wait_tensorcnt(0);
      }
    }
    __syncthreads();   // LDS tiles of `cur` visible to all waves

    const unsigned int* La = lds_a[cur];
    const unsigned int* Lb = lds_b[cur];

    // A fragment (ISA 16-bit A 16x32 layout): two contiguous 16B groups
    FragAB a;
    const int arow = wave * 16 + r;
#pragma unroll
    for (int p = 0; p < 8; ++p)
      a.u[p] = La[arow * 16 + p + ((p >= 4) ? 4 : 0) + (half ? 4 : 0)];

    // 4 WMMAs reuse the A fragment across the 64-wide N tile
#pragma unroll
    for (int nt = 0; nt < 4; ++nt) {
      FragAB b;
      const int base = (nt * 16 + r) * 16 + (half ? 4 : 0);   // two 4-u32 groups (table k order, matches A)
#pragma unroll
      for (int p = 0; p < 8; ++p) b.u[p] = Lb[base + p + ((p >= 4) ? 4 : 0)];
      acc[nt] = __builtin_amdgcn_wmma_f32_16x16x32_bf16(
          false, a.v, false, b.v, (short)0, acc[nt], false, false);
    }
    __syncthreads();   // done reading `cur` before it is overwritten
  }

  // epilogue: D layout — VGPR v holds M = v + 8*half, N = lane&15
#pragma unroll
  for (int nt = 0; nt < 4; ++nt) {
    const int col = bn + nt * 16 + r;
    const float bv = bias ? bias[col] : 0.0f;
#pragma unroll
    for (int v = 0; v < 8; ++v) {
      int rowm = bm + wave * 16 + v + (half ? 8 : 0);
      float x = acc[nt][v] * scale + bv;
      if (relu) x = fmaxf(x, 0.0f);
      if (out_bf16) Cb[(size_t)rowm * N + col] = f32_to_bf16(x);
      else          Cf[(size_t)rowm * N + col] = x;
    }
  }
}

// ---------------- row normalization; emits z (bf16, A-operand) and z-panel (B-operand) ----
__global__ void __launch_bounds__(128)
normalize_rows(const float* __restrict__ zin,
               unsigned short* __restrict__ zb,
               unsigned int* __restrict__ zswz,   // NB x (NP/2) panel
               int M, int D) {
  __shared__ float red[128];
  __shared__ unsigned short zr[128];
  const int i = blockIdx.x, t = threadIdx.x;
  float v = zin[(size_t)i * D + t];
  red[t] = v * v;
  __syncthreads();
  for (int s = 64; s > 0; s >>= 1) {
    if (t < s) red[t] += red[t + s];
    __syncthreads();
  }
  float z = v / sqrtf(red[0]);
  unsigned short b = f32_to_bf16(z);
  zb[(size_t)i * D + t] = b;
  zr[t] = b;
  __syncthreads();
  if (t < (D >> 1))
    zswz[(size_t)i * (D >> 1) + t] = (unsigned)zr[2 * t] | ((unsigned)zr[2 * t + 1] << 16);
}

// ---------------- lse over in-batch negatives (S already has 1/TEMP applied) --------
__global__ void __launch_bounds__(256)
lse_batch_k(const float* __restrict__ S, const int* __restrict__ labels,
            float* __restrict__ out, int n) {
  __shared__ float red[256];
  __shared__ int   redc[256];
  const int i = blockIdx.x, t = threadIdx.x, li = labels[i];
  const float* row = S + (size_t)i * n;
  float m = -INFINITY; int c = 0;
  for (int j = t; j < n; j += 256)
    if (labels[j] != li) { m = fmaxf(m, row[j]); c++; }
  red[t] = m; redc[t] = c;
  __syncthreads();
  for (int s = 128; s > 0; s >>= 1) {
    if (t < s) { red[t] = fmaxf(red[t], red[t + s]); redc[t] += redc[t + s]; }
    __syncthreads();
  }
  m = red[0];
  const int cnt = redc[0];
  __syncthreads();
  if (cnt == 0) { if (t == 0) out[i] = -INFINITY; return; }
  float su = 0.f;
  for (int j = t; j < n; j += 256)
    if (labels[j] != li) su += expf(row[j] - m);
  red[t] = su;
  __syncthreads();
  for (int s = 128; s > 0; s >>= 1) {
    if (t < s) red[t] += red[t + s];
    __syncthreads();
  }
  if (t == 0) out[i] = m + logf(red[0]);
}

// ---------------- exact top-512 radix-select + LSE over queue sims ----------------
__global__ void __launch_bounds__(256)
topk_lse_k(const float* __restrict__ sims, const int* __restrict__ labels,
           const int* __restrict__ qlabels, float* __restrict__ lse_q) {
  __shared__ unsigned int hist[256];
  __shared__ float        redf[256];
  __shared__ unsigned int redu[256];
  __shared__ unsigned int sel[2];

  const int i = blockIdx.x, t = threadIdx.x, li = labels[i];
  const float* row = sims + (size_t)i * NQ;

  unsigned int c = 0; float mx = -INFINITY;
  for (int j = t; j < NQ; j += 256) {
    int q = qlabels[j];
    if (q != li && q >= 0) { c++; mx = fmaxf(mx, row[j]); }
  }
  redu[t] = c; redf[t] = mx;
  __syncthreads();
  for (int s = 128; s > 0; s >>= 1) {
    if (t < s) { redu[t] += redu[t + s]; redf[t] = fmaxf(redf[t], redf[t + s]); }
    __syncthreads();
  }
  const unsigned int oppc = redu[0];
  const float vmax = redf[0];
  __syncthreads();

  if (oppc == 0u) {
    float m = -INFINITY;
    for (int j = t; j < NK; j += 256) m = fmaxf(m, row[j]);
    redf[t] = m; __syncthreads();
    for (int s = 128; s > 0; s >>= 1) { if (t < s) redf[t] = fmaxf(redf[t], redf[t + s]); __syncthreads(); }
    m = redf[0]; __syncthreads();
    float su = 0.f;
    for (int j = t; j < NK; j += 256) su += expf((row[j] - m) * INV_TEMP);
    redf[t] = su; __syncthreads();
    for (int s = 128; s > 0; s >>= 1) { if (t < s) redf[t] += redf[t + s]; __syncthreads(); }
    if (t == 0) lse_q[i] = m * INV_TEMP + logf(redf[0]);
    return;
  }

  if (oppc <= (unsigned)NK) {
    float su = 0.f;
    for (int j = t; j < NQ; j += 256) {
      int q = qlabels[j];
      if (q != li && q >= 0) su += expf((row[j] - vmax) * INV_TEMP);
    }
    redf[t] = su; __syncthreads();
    for (int s = 128; s > 0; s >>= 1) { if (t < s) redf[t] += redf[t + s]; __syncthreads(); }
    if (t == 0) lse_q[i] = vmax * INV_TEMP + logf(redf[0]);
    return;
  }

  unsigned int prefix = 0, need = NK;
  for (int shift = 24; shift >= 0; shift -= 8) {
    hist[t] = 0;
    __syncthreads();
    for (int j = t; j < NQ; j += 256) {
      int q = qlabels[j];
      if (q == li || q < 0) continue;
      unsigned int key = fkey(row[j]);
      if (shift < 24 && (key >> (shift + 8)) != (prefix >> (shift + 8))) continue;
      atomicAdd(&hist[(key >> shift) & 255u], 1u);
    }
    __syncthreads();
    if (t == 0) {
      unsigned int acc2 = 0; int b = 255;
      for (; b >= 0; --b) {
        unsigned int h = hist[b];
        if (acc2 + h >= need) break;
        acc2 += h;
      }
      sel[0] = (unsigned int)b;
      sel[1] = need - acc2;
    }
    __syncthreads();
    prefix |= sel[0] << shift;
    need    = sel[1];
    __syncthreads();
  }

  const unsigned int T = prefix;
  const float tval = funkey(T);
  float su = 0.f;
  for (int j = t; j < NQ; j += 256) {
    int q = qlabels[j];
    if (q == li || q < 0) continue;
    if (fkey(row[j]) > T) su += expf((row[j] - vmax) * INV_TEMP);
  }
  redf[t] = su;
  __syncthreads();
  for (int s = 128; s > 0; s >>= 1) { if (t < s) redf[t] += redf[t + s]; __syncthreads(); }
  if (t == 0) {
    float tot = redf[0] + (float)need * expf((tval - vmax) * INV_TEMP);
    lse_q[i] = vmax * INV_TEMP + logf(tot);
  }
}

// ---------------- final loss reduction ----------------
__global__ void init_acc(float* s, unsigned int* c) { *s = 0.f; *c = 0u; }

__global__ void __launch_bounds__(256)
loss_k(const float* __restrict__ S, const int* __restrict__ labels,
       const float* __restrict__ lse_b, const float* __restrict__ lse_q,
       float* acc_sum, unsigned int* acc_cnt, int n) {
  __shared__ float        red[256];
  __shared__ unsigned int redc[256];
  const int i = blockIdx.x, t = threadIdx.x, li = labels[i];
  const float lneg = logaddexpf_(lse_b[i], lse_q[i]);
  const float* row = S + (size_t)i * n;
  float s = 0.f; unsigned int c = 0;
  for (int j = t; j < n; j += 256) {
    if (j != i && labels[j] == li) {
      float sv = row[j];
      s += logaddexpf_(sv, lneg) - sv;   // = -log_prob
      c++;
    }
  }
  red[t] = s; redc[t] = c;
  __syncthreads();
  for (int st = 128; st > 0; st >>= 1) {
    if (t < st) { red[t] += red[t + st]; redc[t] += redc[t + st]; }
    __syncthreads();
  }
  if (t == 0) { atomicAdd(acc_sum, red[0]); atomicAdd(acc_cnt, redc[0]); }
}

__global__ void finalize_k(const float* acc_sum, const unsigned int* acc_cnt, float* out) {
  unsigned int c = *acc_cnt;
  out[0] = (c > 0u) ? (*acc_sum / (float)c) : 0.0f;
}

// ---------------- host launch ----------------
extern "C" void kernel_launch(void* const* d_in, const int* in_sizes, int n_in,
                              void* d_out, int out_size, void* d_ws, size_t ws_size,
                              hipStream_t stream) {
  (void)in_sizes; (void)n_in; (void)out_size; (void)ws_size;

  const float* emb  = (const float*)d_in[0];
  const int*   lab  = (const int*)d_in[1];
  const float* W1   = (const float*)d_in[2];
  const float* b1   = (const float*)d_in[3];
  const float* W2   = (const float*)d_in[4];
  const float* b2   = (const float*)d_in[5];
  const float* qemb = (const float*)d_in[6];
  const int*   qlab = (const int*)d_in[7];
  float* out = (float*)d_out;

  char* ws = (char*)d_ws;
  size_t off = 0;
  auto take = [&](size_t bytes) -> char* {
    char* p = ws + off;
    off = (off + bytes + 255) & ~(size_t)255;
    return p;
  };

  unsigned short* emb_bf = (unsigned short*)take((size_t)NB * NE * 2);
  unsigned int*   w1s    = (unsigned int*)take((size_t)NE * (NE / 2) * 4);  // panel N=NE, Kp=NE/2
  unsigned int*   w2s    = (unsigned int*)take((size_t)NP * (NE / 2) * 4);  // panel N=NP, Kp=NE/2
  unsigned short* h_bf   = (unsigned short*)take((size_t)NB * NE * 2);
  float*          zpre   = (float*)take((size_t)NB * NP * 4);
  unsigned short* z_bf   = (unsigned short*)take((size_t)NB * NP * 2);
  unsigned int*   zsw    = (unsigned int*)take((size_t)NB * (NP / 2) * 4);  // panel N=NB, Kp=NP/2
  unsigned int*   qsw    = (unsigned int*)take((size_t)NQ * (NP / 2) * 4);  // panel N=NQ, Kp=NP/2
  float*          Smat   = (float*)take((size_t)NB * NB * 4);
  float*          sims   = (float*)take((size_t)NB * NQ * 4);
  float*          lseb   = (float*)take((size_t)NB * 4);
  float*          lseq   = (float*)take((size_t)NB * 4);
  float*          accs   = (float*)take(4);
  unsigned int*   accc   = (unsigned int*)take(4);

  // --- one-time conversions / panel packs ---
  cvt_f32_bf16<<<(NB * NE + 255) / 256, 256, 0, stream>>>(emb, emb_bf, NB * NE);
  pack_B_swz<<<(NE * (NE / 2) + 255) / 256, 256, 0, stream>>>(W1, w1s, NE, NE);
  pack_B_swz<<<(NP * (NE / 2) + 255) / 256, 256, 0, stream>>>(W2, w2s, NE, NP);
  pack_Bt_swz<<<(NQ * (NP / 2) + 255) / 256, 256, 0, stream>>>(qemb, qsw, NQ, NP);

  // --- h = relu(emb @ W1 + b1)  (bf16 out) ---
  gemm_bf16_wmma<<<dim3(NE / 64, NB / 128), 256, 0, stream>>>(
      emb_bf, w1s, b1, nullptr, h_bf, NB, NE, NE, 1.0f, /*relu=*/1, /*bf16=*/1);

  // --- z_pre = h @ W2 + b2  (f32 out) ---
  gemm_bf16_wmma<<<dim3(NP / 64, NB / 128), 256, 0, stream>>>(
      h_bf, w2s, b2, zpre, nullptr, NB, NP, NE, 1.0f, 0, 0);

  // --- normalize rows; emit z (bf16) and z panel ---
  normalize_rows<<<NB, 128, 0, stream>>>(zpre, z_bf, zsw, NB, NP);

  // --- S = (z @ z^T) / TEMP ---
  gemm_bf16_wmma<<<dim3(NB / 64, NB / 128), 256, 0, stream>>>(
      z_bf, zsw, nullptr, Smat, nullptr, NB, NB, NP, INV_TEMP, 0, 0);

  // --- sims_q = z @ Q^T (raw; TEMP applied after top-k) ---
  gemm_bf16_wmma<<<dim3(NQ / 64, NB / 128), 256, 0, stream>>>(
      z_bf, qsw, nullptr, sims, nullptr, NB, NQ, NP, 1.0f, 0, 0);

  // --- reductions ---
  lse_batch_k<<<NB, 256, 0, stream>>>(Smat, lab, lseb, NB);
  topk_lse_k<<<NB, 256, 0, stream>>>(sims, lab, qlab, lseq);

  init_acc<<<1, 1, 0, stream>>>(accs, accc);
  loss_k<<<NB, 256, 0, stream>>>(Smat, lab, lseb, lseq, accs, accc, NB);
  finalize_k<<<1, 1, 0, stream>>>(accs, accc, out);
}
